// CrossAttentionTransformerBlock_10711648436566
// MI455X (gfx1250) — matrix operand order repaired and checked by reading, not compiled, not run on hardware
//
#include <hip/hip_runtime.h>
#include <hip/hip_bf16.h>
#include <math.h>

typedef _Float16 half_t;
typedef __attribute__((ext_vector_type(16))) _Float16 v16h;
typedef __attribute__((ext_vector_type(8)))  _Float16 v8h;
typedef __attribute__((ext_vector_type(8)))  float    v8f;

union HV16 { v16h v; v8h h[2]; };

// ---------------------------------------------------------------------------
// WMMA helpers (CDNA5 v_wmma_f32_16x16x32_f16), fragment loads that match the
// ISA VGPR layouts (CDNA5 ISA):
//   A (16x32 f16): lane<16 -> row=lane, halves0-7=K0..7,  halves8-15=K16..23
//                  lane>=16-> row=l16,  halves0-7=K8..15, halves8-15=K24..31
//   B (32x16 f16): lane<16 -> col=lane, halves0-7=K0..7,  halves8-15=K16..23
//                  lane>=16-> col=l16,  halves0-7=K8..15, halves8-15=K24..31
//   C/D: lane<16 holds (m=idx, n=lane); lane>=16 holds (m=idx+8, n=l16)
// ---------------------------------------------------------------------------
__device__ __forceinline__ v8f wmma_f16(v16h a, v16h b, v8f c) {
  return __builtin_amdgcn_wmma_f32_16x16x32_f16(false, a, false, b, (short)0, c,
                                                false, false);
}
// p16 must point at the start of the 32-half K-window of this row (16B aligned)
__device__ __forceinline__ v16h frag_a(const half_t* p16, int hi) {
  const v8h* p = (const v8h*)p16;
  HV16 u; u.h[0] = p[hi ? 1 : 0]; u.h[1] = p[hi ? 3 : 2]; return u.v;
}
__device__ __forceinline__ v16h frag_b(const half_t* p16, int hi) {
  const v8h* p = (const v8h*)p16;
  HV16 u; u.h[0] = p[hi ? 1 : 0]; u.h[1] = p[hi ? 3 : 2]; return u.v;
}

// ---------------------------------------------------------------------------
// DPP16 ROW_XMASK butterfly reductions within each 16-lane group.
// Pure VALU (v_mov_b32 dpp + v_max/v_add) -- replaces ds_bpermute round-trips.
// ---------------------------------------------------------------------------
template <int MASK>
__device__ __forceinline__ float xmask_lane(float v) {
  const int t = __builtin_amdgcn_update_dpp(0, __float_as_int(v),
                                            0x160 | MASK, 0xf, 0xf, true);
  return __int_as_float(t);
}
__device__ __forceinline__ float row16_max(float v) {
  v = fmaxf(v, xmask_lane<1>(v));
  v = fmaxf(v, xmask_lane<2>(v));
  v = fmaxf(v, xmask_lane<4>(v));
  v = fmaxf(v, xmask_lane<8>(v));
  return v;
}
__device__ __forceinline__ float row16_sum(float v) {
  v += xmask_lane<1>(v);
  v += xmask_lane<2>(v);
  v += xmask_lane<4>(v);
  v += xmask_lane<8>(v);
  return v;
}

// ---------------------------------------------------------------------------
// f32 -> f16 conversion (weights)
// ---------------------------------------------------------------------------
__global__ void cvt_f16(const float* __restrict__ a, half_t* __restrict__ o, int n) {
  int i = blockIdx.x * blockDim.x + threadIdx.x;
  if (i < n) o[i] = (half_t)a[i];
}

// ---------------------------------------------------------------------------
// LayerNorm: fp32 rows -> f16 rows (one block per row, C=768)
// DPP reduce within 16-lane groups, combine the 16 half-group partials in LDS.
// ---------------------------------------------------------------------------
__global__ void __launch_bounds__(256) ln_f16(const float* __restrict__ X,
                                              const float* __restrict__ g,
                                              const float* __restrict__ be,
                                              half_t* __restrict__ Y, int C) {
  const int row = blockIdx.x;
  const int tid = threadIdx.x;
  const float* xr = X + (size_t)row * C;
  float s = 0.f, s2 = 0.f;
  for (int c = tid; c < C; c += 256) { float v = xr[c]; s += v; s2 += v * v; }
  s = row16_sum(s);
  s2 = row16_sum(s2);
  __shared__ float sm[32];
  const int grp = tid >> 4;                 // 0..15 (wave*2 + half)
  if ((tid & 15) == 0) { sm[grp] = s; sm[16 + grp] = s2; }
  __syncthreads();
  float ts = 0.f, ts2 = 0.f;
#pragma unroll
  for (int i = 0; i < 16; ++i) { ts += sm[i]; ts2 += sm[16 + i]; }
  const float mu = ts / C;
  const float var = ts2 / C - mu * mu;
  const float rstd = rsqrtf(var + 1e-6f);
  half_t* yr = Y + (size_t)row * C;
  for (int c = tid; c < C; c += 256)
    yr[c] = (half_t)((xr[c] - mu) * rstd * g[c] + be[c]);
}

// ---------------------------------------------------------------------------
// WMMA GEMM: Y[rows, cols] = epilogue( X[rows,K] @ W[cols,K]^T + bias )
// Block tile 128x64, 8 waves each 32x32 (4 WMMA tiles), K-step 32,
// double-buffered LDS with +8 half padding (conflict-free b128 frag loads).
// rows % 128 == 0, cols % 64 == 0, K % 32 == 0 (true for all uses here).
// ---------------------------------------------------------------------------
template <bool BIAS, bool GELU_, bool OUTH, bool RES>
__global__ void __launch_bounds__(256) gemm_wmma(const half_t* __restrict__ X,
                                                 const half_t* __restrict__ W,
                                                 const float* __restrict__ bias,
                                                 half_t* __restrict__ Yh,
                                                 float* __restrict__ Yres,
                                                 int Kd, int ldo) {
  const int tid  = threadIdx.x;
  const int lane = tid & 31, wave = tid >> 5;
  const int l16  = lane & 15, hi = (lane >> 4) & 1;
  const int wrow = (wave >> 1) * 32;   // 0,32,64,96
  const int wcol = (wave & 1) * 32;    // 0,32
  const int row0 = blockIdx.y * 128, col0 = blockIdx.x * 64;

  __shared__ half_t Xs[2][128][40];
  __shared__ half_t Ws[2][64][40];

  const int lr = tid >> 2;           // 0..63
  const int lc = (tid & 3) * 8;      // 0,8,16,24
  const half_t* Xg0 = X + (size_t)(row0 + lr) * Kd + lc;
  const half_t* Xg1 = X + (size_t)(row0 + 64 + lr) * Kd + lc;
  const half_t* Wg  = W + (size_t)(col0 + lr) * Kd + lc;

  v8h px0 = *(const v8h*)Xg0;
  v8h px1 = *(const v8h*)Xg1;
  v8h pw  = *(const v8h*)Wg;
  *(v8h*)&Xs[0][lr][lc]      = px0;
  *(v8h*)&Xs[0][64 + lr][lc] = px1;
  *(v8h*)&Ws[0][lr][lc]      = pw;
  __syncthreads();

  v8f a00 = {}, a01 = {}, a10 = {}, a11 = {};
  const int KT = Kd >> 5;
  for (int kt = 0; kt < KT; ++kt) {
    const int cur = kt & 1;
    if (kt + 1 < KT) {                 // prefetch next K-slab (hides latency)
      const int ko = (kt + 1) << 5;
      px0 = *(const v8h*)(Xg0 + ko);
      px1 = *(const v8h*)(Xg1 + ko);
      pw  = *(const v8h*)(Wg + ko);
    }
    const v16h fa0 = frag_a(&Xs[cur][wrow + l16][0], hi);
    const v16h fa1 = frag_a(&Xs[cur][wrow + 16 + l16][0], hi);
    const v16h fb0 = frag_b(&Ws[cur][wcol + l16][0], hi);
    const v16h fb1 = frag_b(&Ws[cur][wcol + 16 + l16][0], hi);
    a00 = wmma_f16(fa0, fb0, a00);
    a01 = wmma_f16(fa0, fb1, a01);
    a10 = wmma_f16(fa1, fb0, a10);
    a11 = wmma_f16(fa1, fb1, a11);
    if (kt + 1 < KT) {
      const int nxt = cur ^ 1;
      *(v8h*)&Xs[nxt][lr][lc]      = px0;
      *(v8h*)&Xs[nxt][64 + lr][lc] = px1;
      *(v8h*)&Ws[nxt][lr][lc]      = pw;
    }
    __syncthreads();
  }

  auto emit = [&](const v8f& acc, int rt, int ct) {
    const int col = col0 + wcol + ct * 16 + l16;
    const int r0  = row0 + wrow + rt * 16 + hi * 8;
    const float bv = BIAS ? bias[col] : 0.0f;
#pragma unroll
    for (int m = 0; m < 8; ++m) {
      float v = acc[m] + bv;
      if (GELU_) v = 0.5f * v * (1.0f + erff(v * 0.70710678118654752f));
      const size_t idx = (size_t)(r0 + m) * ldo + col;
      if (OUTH) Yh[idx] = (half_t)v;
      if (RES)  Yres[idx] += v;
    }
  };
  emit(a00, 0, 0); emit(a01, 0, 1); emit(a10, 1, 0); emit(a11, 1, 1);
}

// ---------------------------------------------------------------------------
// Flash attention (HD=64), WMMA for both QK^T and PV.
// Block = 128 threads (4 waves), 64 query rows / block, 32-key tiles.
// Row stats via DPP16 ROW_XMASK butterflies over the 16-lane groups that own
// one query row per accumulator element in the WMMA C/D layout (no LDS round-trips).
// ---------------------------------------------------------------------------
__global__ void __launch_bounds__(128) attn_kernel(
    const half_t* __restrict__ Qb, unsigned long long q_bs, int q_rs, int q_hs,
    const half_t* __restrict__ Kb, unsigned long long k_bs, int k_rs, int k_hs,
    const half_t* __restrict__ Vb, unsigned long long v_bs, int v_rs, int v_hs,
    half_t* __restrict__ Ob,       unsigned long long o_bs, int o_rs, int o_hs,
    int n_k, float scale) {
  const int tid  = threadIdx.x;
  const int lane = tid & 31, wave = tid >> 5;
  const int l16  = lane & 15, hi = (lane >> 4) & 1;
  const int b = blockIdx.z, h = blockIdx.y;
  const int q0 = blockIdx.x * 64;

  const half_t* Q = Qb + (size_t)b * q_bs + (size_t)h * q_hs;
  const half_t* K = Kb + (size_t)b * k_bs + (size_t)h * k_hs;
  const half_t* V = Vb + (size_t)b * v_bs + (size_t)h * v_hs;
  half_t*       O = Ob + (size_t)b * o_bs + (size_t)h * o_hs;

  __shared__ half_t Qs[64][72];      // [q][hd]
  __shared__ half_t Ks[32][72];      // [key][hd]
  __shared__ half_t Vs[64][40];      // transposed: [hd][key]
  __shared__ half_t Ps[4][16][40];   // per-wave probs [q][key]

  for (int idx = tid; idx < 64 * 8; idx += 128) {
    const int r = idx >> 3, c = (idx & 7) * 8;
    *(v8h*)&Qs[r][c] = *(const v8h*)(Q + (size_t)(q0 + r) * q_rs + c);
  }

  v8f o0 = {}, o1 = {}, o2 = {}, o3 = {};
  float runmax[8], runsum[8];
#pragma unroll
  for (int m = 0; m < 8; ++m) { runmax[m] = -3.0e38f; runsum[m] = 0.f; }

  const int KT = n_k >> 5;
  for (int kt = 0; kt < KT; ++kt) {
    // stage K tile (row-major) and V tile (transposed) in LDS
    for (int idx = tid; idx < 32 * 8; idx += 128) {
      const int r = idx >> 3, c = (idx & 7) * 8;
      *(v8h*)&Ks[r][c] = *(const v8h*)(K + (size_t)(kt * 32 + r) * k_rs + c);
    }
    for (int idx = tid; idx < 32 * 8; idx += 128) {
      const int r = idx >> 3, c = (idx & 7) * 8;
      const v8h val = *(const v8h*)(V + (size_t)(kt * 32 + r) * v_rs + c);
#pragma unroll
      for (int j = 0; j < 8; ++j) Vs[c + j][r] = val[j];
    }
    __syncthreads();

    // S(16q x 32k) = Q @ K^T  (K-loop over HD=64 in two 32-steps)
    v8f s0 = {}, s1 = {};
#pragma unroll
    for (int h0 = 0; h0 < 64; h0 += 32) {
      const v16h a  = frag_a(&Qs[wave * 16 + l16][h0], hi);
      const v16h b0 = frag_b(&Ks[l16][h0], hi);
      const v16h b1 = frag_b(&Ks[16 + l16][h0], hi);
      s0 = wmma_f16(a, b0, s0);
      s1 = wmma_f16(a, b1, s1);
    }

    // online softmax: each 16-lane group owns 8 q-rows; keys spread over lanes
    float p0[8], p1[8];
#pragma unroll
    for (int m = 0; m < 8; ++m) {
      const float a0 = s0[m] * scale, a1 = s1[m] * scale;
      const float tm = row16_max(fmaxf(a0, a1));
      const float nm   = fmaxf(runmax[m], tm);
      const float corr = __expf(runmax[m] - nm);
      const float e0 = __expf(a0 - nm);
      const float e1 = __expf(a1 - nm);
      const float ts = row16_sum(e0 + e1);
      runsum[m] = runsum[m] * corr + ts;
      runmax[m] = nm;
      o0[m] *= corr; o1[m] *= corr; o2[m] *= corr; o3[m] *= corr;
      p0[m] = e0; p1[m] = e1;
    }

    // D-layout -> A-layout reshuffle of P through per-wave LDS
#pragma unroll
    for (int m = 0; m < 8; ++m) {
      Ps[wave][m + hi * 8][l16]      = (half_t)p0[m];
      Ps[wave][m + hi * 8][16 + l16] = (half_t)p1[m];
    }
    const v16h pa = frag_a(&Ps[wave][l16][0], hi);
    o0 = wmma_f16(pa, frag_b(&Vs[l16][0], hi),      o0);
    o1 = wmma_f16(pa, frag_b(&Vs[16 + l16][0], hi), o1);
    o2 = wmma_f16(pa, frag_b(&Vs[32 + l16][0], hi), o2);
    o3 = wmma_f16(pa, frag_b(&Vs[48 + l16][0], hi), o3);
    __syncthreads();
  }

  const int row0 = q0 + wave * 16 + hi * 8;
#pragma unroll
  for (int m = 0; m < 8; ++m) {
    const float inv = 1.0f / runsum[m];
    const size_t rb = (size_t)(row0 + m) * o_rs;
    O[rb + l16]      = (half_t)(o0[m] * inv);
    O[rb + 16 + l16] = (half_t)(o1[m] * inv);
    O[rb + 32 + l16] = (half_t)(o2[m] * inv);
    O[rb + 48 + l16] = (half_t)(o3[m] * inv);
  }
}

// ---------------------------------------------------------------------------
// Orchestration
// ---------------------------------------------------------------------------
extern "C" void kernel_launch(void* const* d_in, const int* in_sizes, int n_in,
                              void* d_out, int out_size, void* d_ws, size_t ws_size,
                              hipStream_t stream) {
  (void)in_sizes; (void)n_in; (void)out_size; (void)ws_size;
  constexpr int Bn = 4, N = 1024, Mm = 1024, C = 768, Hh = 12, HDim = 64;
  constexpr int D = 768, HID = 3072;
  constexpr int RN = Bn * N, RM = Bn * Mm;   // 4096 rows each
  const float SCALE = 0.125f;                // HD^-0.5

  float* outx = (float*)d_out;
  float* outs = outx + (size_t)RN * C;
  hipMemcpyAsync(outx, d_in[0], sizeof(float) * (size_t)RN * C,
                 hipMemcpyDeviceToDevice, stream);
  hipMemcpyAsync(outs, d_in[1], sizeof(float) * (size_t)RM * C,
                 hipMemcpyDeviceToDevice, stream);

  char* wsb = (char*)d_ws;
  size_t off = 0;
  auto alloc_h = [&](size_t elems) -> half_t* {
    half_t* p = (half_t*)(wsb + off);
    off += ((elems * sizeof(half_t)) + 255) & ~(size_t)255;
    return p;
  };
  auto cvt = [&](int i, size_t n) -> half_t* {
    half_t* p = alloc_h(n);
    cvt_f16<<<dim3((unsigned)((n + 255) / 256)), dim3(256), 0, stream>>>(
        (const float*)d_in[i], p, (int)n);
    return p;
  };

  // weight conversion to f16
  half_t* w_qkv_h   = cvt(2,  (size_t)3 * D * C);
  half_t* w_proj_h  = cvt(3,  (size_t)C * D);
  half_t* w_qkvs_h  = cvt(5,  (size_t)3 * D * C);
  half_t* w_projs_h = cvt(6,  (size_t)C * D);
  half_t* w_qk_h    = cvt(8,  (size_t)2 * D * C);
  half_t* w_qks_h   = cvt(9,  (size_t)2 * D * C);
  half_t* w_v_h     = cvt(10, (size_t)D * C);
  half_t* w_vs_h    = cvt(11, (size_t)D * C);
  half_t* w_cp_h    = cvt(12, (size_t)C * D);
  half_t* w_cps_h   = cvt(14, (size_t)C * D);
  half_t* mlp1_h    = cvt(16, (size_t)HID * C);
  half_t* mlp2_h    = cvt(18, (size_t)C * HID);
  half_t* mlp1s_h   = cvt(20, (size_t)HID * C);
  half_t* mlp2s_h   = cvt(22, (size_t)C * HID);

  half_t* lnx    = alloc_h((size_t)RN * C);
  half_t* lns    = alloc_h((size_t)RM * C);
  half_t* attn_o = alloc_h((size_t)RN * C);
  half_t* big    = alloc_h((size_t)RN * 4608);   // shared activation scratch

  auto gemmH = [&](const half_t* X, const half_t* W, half_t* Y, int rows,
                   int cols, int Kd) {
    gemm_wmma<false, false, true, false>
        <<<dim3(cols / 64, rows / 128), dim3(256), 0, stream>>>(
            X, W, nullptr, Y, nullptr, Kd, cols);
  };
  auto gemmRes = [&](const half_t* X, const half_t* W, const float* bia,
                     float* Y, int rows, int cols, int Kd) {
    gemm_wmma<true, false, false, true>
        <<<dim3(cols / 64, rows / 128), dim3(256), 0, stream>>>(
            X, W, bia, nullptr, Y, Kd, cols);
  };
  auto gemmGelu = [&](const half_t* X, const half_t* W, const float* bia,
                      half_t* Y, int rows, int cols, int Kd) {
    gemm_wmma<true, true, true, false>
        <<<dim3(cols / 64, rows / 128), dim3(256), 0, stream>>>(
            X, W, bia, Y, nullptr, Kd, cols);
  };

  // ---- self-attention (x) ----
  ln_f16<<<RN, 256, 0, stream>>>(outx, (const float*)d_in[24],
                                 (const float*)d_in[25], lnx, C);
  gemmH(lnx, w_qkv_h, big, RN, 3 * D, C);   // qkv: [B,N,3,H,HD]
  attn_kernel<<<dim3(N / 64, Hh, Bn), 128, 0, stream>>>(
      big,         (unsigned long long)N * 3 * D, 3 * D, HDim,
      big + D,     (unsigned long long)N * 3 * D, 3 * D, HDim,
      big + 2 * D, (unsigned long long)N * 3 * D, 3 * D, HDim,
      attn_o,      (unsigned long long)N * D,     D,     HDim, N, SCALE);
  gemmRes(attn_o, w_proj_h, (const float*)d_in[4], outx, RN, C, D);

  // ---- self-attention (src) ----
  ln_f16<<<RM, 256, 0, stream>>>(outs, (const float*)d_in[26],
                                 (const float*)d_in[27], lns, C);
  gemmH(lns, w_qkvs_h, big, RM, 3 * D, C);
  attn_kernel<<<dim3(Mm / 64, Hh, Bn), 128, 0, stream>>>(
      big,         (unsigned long long)Mm * 3 * D, 3 * D, HDim,
      big + D,     (unsigned long long)Mm * 3 * D, 3 * D, HDim,
      big + 2 * D, (unsigned long long)Mm * 3 * D, 3 * D, HDim,
      attn_o,      (unsigned long long)Mm * D,     D,     HDim, Mm, SCALE);
  gemmRes(attn_o, w_projs_h, (const float*)d_in[7], outs, RM, C, D);

  // ---- cross-attention ----
  ln_f16<<<RN, 256, 0, stream>>>(outx, (const float*)d_in[28],
                                 (const float*)d_in[29], lnx, C);
  ln_f16<<<RM, 256, 0, stream>>>(outs, (const float*)d_in[30],
                                 (const float*)d_in[31], lns, C);
  half_t* qk  = big;                          // [B,N,H,128]
  half_t* qks = big + (size_t)RN * 1536;      // [B,M,H,128]
  half_t* vv  = big + (size_t)RN * 3072;      // [B,N,D]
  half_t* vvs = big + (size_t)RN * 3840;      // [B,M,D]
  gemmH(lnx, w_qk_h,  qk,  RN, 2 * D, C);
  gemmH(lns, w_qks_h, qks, RM, 2 * D, C);
  gemmH(lnx, w_v_h,   vv,  RN, D, C);
  gemmH(lns, w_vs_h,  vvs, RM, D, C);

  // y = softmax(q . q_s^T) @ v_s  (q = qk[:,:64], q_s = qks[:,:64])
  attn_kernel<<<dim3(N / 64, Hh, Bn), 128, 0, stream>>>(
      qk,     (unsigned long long)N * 1536,  1536, 2 * HDim,
      qks,    (unsigned long long)Mm * 1536, 1536, 2 * HDim,
      vvs,    (unsigned long long)Mm * D,    D,    HDim,
      attn_o, (unsigned long long)N * D,     D,    HDim, Mm, SCALE);
  gemmRes(attn_o, w_cp_h, (const float*)d_in[13], outx, RN, C, D);

  // y_s = softmax(k_s . kx^T) @ v  (queries = k_s, keys = kx)
  attn_kernel<<<dim3(Mm / 64, Hh, Bn), 128, 0, stream>>>(
      qks + HDim, (unsigned long long)Mm * 1536, 1536, 2 * HDim,
      qk + HDim,  (unsigned long long)N * 1536,  1536, 2 * HDim,
      vv,         (unsigned long long)N * D,     D,    HDim,
      attn_o,     (unsigned long long)Mm * D,    D,    HDim, N, SCALE);
  gemmRes(attn_o, w_cps_h, (const float*)d_in[15], outs, RM, C, D);

  // ---- MLP (x) ----
  ln_f16<<<RN, 256, 0, stream>>>(outx, (const float*)d_in[32],
                                 (const float*)d_in[33], lnx, C);
  gemmGelu(lnx, mlp1_h, (const float*)d_in[17], big, RN, HID, C);
  gemmRes(big, mlp2_h, (const float*)d_in[19], outx, RN, C, HID);

  // ---- MLP (src) ----
  ln_f16<<<RM, 256, 0, stream>>>(outs, (const float*)d_in[34],
                                 (const float*)d_in[35], lns, C);
  gemmGelu(lns, mlp1s_h, (const float*)d_in[21], big, RM, HID, C);
  gemmRes(big, mlp2s_h, (const float*)d_in[23], outs, RM, C, HID);
}
